// BDH_4406636445711
// MI455X (gfx1250) — compile-verified, matrix operand order checked by reading, not run on hardware
//
#include <hip/hip_runtime.h>

// ---------------- model constants ----------------
constexpr int Bc = 2, Tc = 2048, Hc = 4, Dc = 256, Kc = 1024, Nc = 4096, Vc = 32000, Lc = 6;
constexpr int Mc = Bc * Tc; // 4096 rows

typedef __attribute__((ext_vector_type(16))) __bf16 v16bf;
typedef __attribute__((ext_vector_type(8)))  float  v8f;

union BF16Frag { v16bf v; unsigned short u[16]; uint4 q[2]; };

static __device__ __forceinline__ unsigned short f2bf(float f) {
    union { float f; unsigned int u; } x; x.f = f;
    unsigned int r = x.u + 0x7FFFu + ((x.u >> 16) & 1u);  // round-to-nearest-even
    return (unsigned short)(r >> 16);
}

// ---------------- elementwise helpers ----------------
__global__ void cvt_f32_bf16(const float* __restrict__ in, unsigned short* __restrict__ out, int n) {
    int i = blockIdx.x * 256 + threadIdx.x;
    if (i < n) out[i] = f2bf(in[i]);
}

// batched transpose + convert: in [z][R][C] f32 -> out [z][C][R] bf16
__global__ void cvt_tr_f32_bf16(const float* __restrict__ in, unsigned short* __restrict__ out,
                                int R, int C, int total) {
    int i = blockIdx.x * 256 + threadIdx.x;
    if (i >= total) return;
    int rc = R * C;
    int z = i / rc, rem = i - z * rc;
    int r = rem / C, c = rem - r * C;
    out[(size_t)z * rc + (size_t)c * R + r] = f2bf(in[i]);
}

static __device__ __forceinline__ float block_sum256(float v, float* sm) {
    int t = threadIdx.x;
    sm[t] = v; __syncthreads();
#pragma unroll
    for (int s = 128; s > 0; s >>= 1) { if (t < s) sm[t] += sm[t + s]; __syncthreads(); }
    float r = sm[0]; __syncthreads();
    return r;
}

// v = ln(emb[input]) , one row (D=256) per 256-thread block
__global__ void embed_ln(const int* __restrict__ idx, const float* __restrict__ emb,
                         float* __restrict__ v) {
    __shared__ float sm[256];
    int row = blockIdx.x, t = threadIdx.x;
    int tok = idx[row];
    float x = emb[(size_t)tok * Dc + t];
    float mu  = block_sum256(x, sm) * (1.f / Dc);
    float d   = x - mu;
    float var = block_sum256(d * d, sm) * (1.f / Dc);
    v[(size_t)row * Dc + t] = d * rsqrtf(var + 1e-5f);
}

// v += pos ; emit bf16 copy (row-major) and bf16 transposed copy per batch [b][d][t]
__global__ void addpos_cvt(float* __restrict__ v, const float* __restrict__ pos,
                           unsigned short* __restrict__ vp, unsigned short* __restrict__ vpT) {
    int i = blockIdx.x * 256 + threadIdx.x;       // over Mc*Dc
    int m = i >> 8, d = i & 255;
    int t = m & (Tc - 1);
    int b = m >> 11;
    float x = v[i] + pos[t * Dc + d];
    v[i] = x;
    unsigned short h = f2bf(x);
    vp[i] = h;
    vpT[(size_t)b * Dc * Tc + (size_t)d * Tc + t] = h;
}

// RoPE on x (already ReLU'd by GEMM epilogue); write q (row-major) and qT ([z][k][t])
__global__ void rope_q(const float* __restrict__ x, unsigned short* __restrict__ q,
                       unsigned short* __restrict__ qT) {
    int i = blockIdx.x * 256 + threadIdx.x;       // over B*H*T*K
    int k = i & (Kc - 1);
    int t = (i >> 10) & (Tc - 1);
    int z = i >> 21;                               // (b*H + h)
    int fi = k & 511;
    // inv_freq = 10000^(-2*fi/1024) = exp(-fi * 2*ln(10000)/1024)
    float invf = __expf(-0.017988946f * (float)fi);
    float ang = (float)t * invf;
    float c, s;
    __sincosf(ang, &s, &c);
    float xk = x[i];
    float rh = (k < 512) ? -x[i + 512] : x[i - 512];
    unsigned short hv = f2bf(xk * c + rh * s);
    q[i] = hv;
    qT[(size_t)z * Tc * Kc + (size_t)k * Tc + t] = hv;
}

// row LayerNorm over D=256, bf16 out (for ln(a))
__global__ void ln_rows_bf16(const float* __restrict__ a, unsigned short* __restrict__ o) {
    __shared__ float sm[256];
    int row = blockIdx.x, t = threadIdx.x;
    float x = a[(size_t)row * Dc + t];
    float mu  = block_sum256(x, sm) * (1.f / Dc);
    float d   = x - mu;
    float var = block_sum256(d * d, sm) * (1.f / Dc);
    o[(size_t)row * Dc + t] = f2bf(d * rsqrtf(var + 1e-5f));
}

// v = ln(v + ln(w)) in place
__global__ void residual_ln(const float* __restrict__ w, float* __restrict__ v) {
    __shared__ float sm[256];
    int row = blockIdx.x, t = threadIdx.x;
    size_t i = (size_t)row * Dc + t;
    float x = w[i];
    float mu   = block_sum256(x, sm) * (1.f / Dc);
    float d    = x - mu;
    float var  = block_sum256(d * d, sm) * (1.f / Dc);
    float lw   = d * rsqrtf(var + 1e-5f);
    float y    = v[i] + lw;
    float mu2  = block_sum256(y, sm) * (1.f / Dc);
    float d2   = y - mu2;
    float var2 = block_sum256(d2 * d2, sm) * (1.f / Dc);
    v[i] = d2 * rsqrtf(var2 + 1e-5f);
}

// ---------------- generic bf16 WMMA GEMM ----------------
// C[M,N] = A[M,Kd] * BT[N,Kd]^T ; A row-major (lda), B given TRANSPOSED row-major (ldbt).
// Both operands are staged k-contiguous in LDS so every WMMA fragment is two aligned
// 16-byte ds_load_b128's (ISA 7.12.2: element e -> k = (e<8?e:e+8) + (lane>>4)*8).
// C addressing: addr = row*ldc + col*ccs (ccs!=1 lets an epilogue store transposed).
// blockIdx.z decomposed as (zb,zh) = (z/HH, z%HH) with per-index slice offsets.
// EPI: 0 = f32 out, 1 = relu f32 out, 2 = bf16 out, 3 = relu(acc)*gate -> bf16 out.
constexpr int BMt = 128, BNt = 128, BKt = 32;
constexpr int LDAS = BKt + 8;   // 40 ushorts = 80 B row stride: 16B-aligned, banks spread

template <int EPI>
__global__ __launch_bounds__(256) void wmma_gemm(
    const unsigned short* __restrict__ A, const unsigned short* __restrict__ BT,
    void* __restrict__ Cv, const float* __restrict__ G,
    int M, int N, int Kd,
    int lda, int ldbt, int ldc, int ccs, int ldg,
    long long aSb, long long aSh, long long bSb, long long bSh,
    long long cSb, long long cSh, long long gSb, long long gSh, int HH) {
    __shared__ __align__(16) unsigned short As[BMt * LDAS];
    __shared__ __align__(16) unsigned short Bs[BNt * LDAS];

    int zb = blockIdx.z / HH, zh = blockIdx.z % HH;
    const unsigned short* Ab = A  + (size_t)(zb * aSb + zh * aSh);
    const unsigned short* Bb = BT + (size_t)(zb * bSb + zh * bSh);

    int tid = threadIdx.x;
    int lane = tid & 31, wv = tid >> 5;
    int l16 = lane & 15, lh = lane >> 4;
    int wM = wv >> 1, wN = wv & 1;       // waves: 4 in M x 2 in N
    int m0 = wM * 32, n0 = wN * 64;      // wave tile: 32x64 = (2 x 4) 16x16 tiles

    v8f acc[2][4] = {};

    int nk = Kd / BKt;
    long long mBase = (long long)blockIdx.y * BMt;
    long long nBase = (long long)blockIdx.x * BNt;

    // staging coords: 512 16-byte chunks per tile, 2 per thread
    int r0 = tid >> 2;                 // 0..63  (row/col index, +64 for second chunk)
    int kp = (tid & 3) * 8;            // k offset within 32-wide tile slab, ushorts

    for (int kb = 0; kb < nk; ++kb) {
        long long kOff = (long long)kb * BKt + kp;
#pragma unroll
        for (int i = 0; i < 2; i++) {
            int r = r0 + 64 * i;
            *(uint4*)&As[r * LDAS + kp] =
                *(const uint4*)&Ab[(size_t)((mBase + r) * (long long)lda + kOff)];
        }
#pragma unroll
        for (int i = 0; i < 2; i++) {
            int c = r0 + 64 * i;
            *(uint4*)&Bs[c * LDAS + kp] =
                *(const uint4*)&Bb[(size_t)((nBase + c) * (long long)ldbt + kOff)];
        }
        __syncthreads();

        BF16Frag af[2], bfr[4];
#pragma unroll
        for (int mt = 0; mt < 2; mt++) {
            int r = m0 + mt * 16 + l16;
            af[mt].q[0] = *(const uint4*)&As[r * LDAS + lh * 8];
            af[mt].q[1] = *(const uint4*)&As[r * LDAS + 16 + lh * 8];
        }
#pragma unroll
        for (int nt = 0; nt < 4; nt++) {
            int c = n0 + nt * 16 + l16;
            bfr[nt].q[0] = *(const uint4*)&Bs[c * LDAS + lh * 8];
            bfr[nt].q[1] = *(const uint4*)&Bs[c * LDAS + 16 + lh * 8];
        }
#pragma unroll
        for (int mt = 0; mt < 2; mt++)
#pragma unroll
            for (int nt = 0; nt < 4; nt++)
                acc[mt][nt] = __builtin_amdgcn_wmma_f32_16x16x32_bf16(
                    false, af[mt].v, false, bfr[nt].v, (short)0, acc[mt][nt], false, false);
        __syncthreads();
    }

    // epilogue (C/D layout: col = l16, row = r + lh*8 within a 16x16 tile)
    size_t cOff = (size_t)(zb * cSb + zh * cSh);
    size_t gOff = (size_t)(zb * gSb + zh * gSh);
    float* Cf = (float*)Cv;
    unsigned short* Cb = (unsigned short*)Cv;
#pragma unroll
    for (int mt = 0; mt < 2; mt++) {
#pragma unroll
        for (int nt = 0; nt < 4; nt++) {
#pragma unroll
            for (int r = 0; r < 8; r++) {
                long long row = mBase + m0 + mt * 16 + lh * 8 + r;
                long long col = nBase + n0 + nt * 16 + l16;
                float val = acc[mt][nt][r];
                size_t ci = cOff + (size_t)(row * (long long)ldc + col * (long long)ccs);
                if (EPI == 0) {
                    Cf[ci] = val;
                } else if (EPI == 1) {
                    Cf[ci] = val > 0.f ? val : 0.f;
                } else if (EPI == 2) {
                    Cb[ci] = f2bf(val);
                } else {
                    float g = G[gOff + (size_t)(row * (long long)ldg + col)];
                    float rv = val > 0.f ? val : 0.f;
                    Cb[ci] = f2bf(rv * g);
                }
            }
        }
    }
}

// ---------------- host orchestration ----------------
extern "C" void kernel_launch(void* const* d_in, const int* in_sizes, int n_in,
                              void* d_out, int out_size, void* d_ws, size_t ws_size,
                              hipStream_t stream) {
    (void)in_sizes; (void)n_in; (void)out_size; (void)ws_size;
    const int*   input = (const int*)  d_in[0];
    const float* emb   = (const float*)d_in[1];
    const float* pos   = (const float*)d_in[2];
    const float* Dx    = (const float*)d_in[3];
    const float* Dy    = (const float*)d_in[4];
    const float* E     = (const float*)d_in[5];
    const float* ro    = (const float*)d_in[6];
    float* out = (float*)d_out;

    char* p = (char*)d_ws;
    auto take = [&](size_t bytes) -> char* {
        char* r = p; p += (bytes + 255) & ~(size_t)255; return r;
    };

    float*          v    = (float*)take((size_t)Mc * Dc * 4);
    unsigned short* vp   = (unsigned short*)take((size_t)Mc * Dc * 2);
    unsigned short* vpT  = (unsigned short*)take((size_t)Mc * Dc * 2);  // [b][d][t]
    unsigned short* DxT  = (unsigned short*)take((size_t)Hc * Dc * Kc * 2); // [h][k][d]
    unsigned short* DyT  = (unsigned short*)take((size_t)Hc * Dc * Kc * 2); // [h][k][d]
    unsigned short* ET   = (unsigned short*)take((size_t)Nc * Dc * 2);      // [d][n]
    unsigned short* roT  = (unsigned short*)take((size_t)Dc * Vc * 2);      // [v][d]
    float*          x    = (float*)take((size_t)Bc * Hc * Tc * Kc * 4);
    unsigned short* qb   = (unsigned short*)take((size_t)Bc * Hc * Tc * Kc * 2); // [z][t][k]
    unsigned short* qT   = (unsigned short*)take((size_t)Bc * Hc * Tc * Kc * 2); // [z][k][t]
    unsigned short* uT   = (unsigned short*)take((size_t)Bc * Hc * Kc * Dc * 2); // [z][d][k]
    float*          a    = (float*)take((size_t)Bc * Hc * Tc * Dc * 4);
    unsigned short* lab  = (unsigned short*)take((size_t)Bc * Hc * Tc * Dc * 2);
    unsigned short* yb   = (unsigned short*)take((size_t)Bc * Tc * Nc * 2);
    float*          w    = (float*)take((size_t)Mc * Dc * 4);

    // weights -> transposed bf16 (one-time per launch; deterministic)
    cvt_tr_f32_bf16<<<(Hc * Dc * Kc) / 256, 256, 0, stream>>>(Dx, DxT, Dc, Kc, Hc * Dc * Kc);
    cvt_tr_f32_bf16<<<(Hc * Dc * Kc) / 256, 256, 0, stream>>>(Dy, DyT, Dc, Kc, Hc * Dc * Kc);
    cvt_tr_f32_bf16<<<(Nc * Dc) / 256, 256, 0, stream>>>(E, ET, Nc, Dc, Nc * Dc);
    cvt_tr_f32_bf16<<<(Dc * Vc) / 256, 256, 0, stream>>>(ro, roT, Dc, Vc, Dc * Vc);

    embed_ln<<<Mc, 256, 0, stream>>>(input, emb, v);

    for (int l = 0; l < Lc; l++) {
        addpos_cvt<<<(Mc * Dc) / 256, 256, 0, stream>>>(v, pos, vp, vpT);

        // x = relu(v' @ Dx[h]) : per (b,h), M=T, N=K, Kd=D, f32 out
        wmma_gemm<1><<<dim3(Kc / BNt, Tc / BMt, Bc * Hc), 256, 0, stream>>>(
            vp, DxT, x, nullptr,
            Tc, Kc, Dc,
            Dc, Dc, Kc, 1, 0,
            (long long)Tc * Dc, 0, 0, (long long)Dc * Kc,
            (long long)Hc * Tc * Kc, (long long)Tc * Kc, 0, 0, Hc);

        rope_q<<<(Bc * Hc * Tc * Kc) / 256, 256, 0, stream>>>(x, qb, qT);

        // u = q^T @ v'  ((q q^T) v == q (q^T v)); A=qT row-major, store transposed -> uT[d][k]
        wmma_gemm<2><<<dim3(Dc / BNt, Kc / BMt, Bc * Hc), 256, 0, stream>>>(
            qT, vpT, uT, nullptr,
            Kc, Dc, Tc,
            Tc, Tc, 1, Kc, 0,
            (long long)Hc * Tc * Kc, (long long)Tc * Kc, (long long)Tc * Dc, 0,
            (long long)Hc * Kc * Dc, (long long)Kc * Dc, 0, 0, Hc);

        // a = q @ u : f32 out [T,D]   (B operand is uT[d][k], i.e. u transposed: perfect)
        wmma_gemm<0><<<dim3(Dc / BNt, Tc / BMt, Bc * Hc), 256, 0, stream>>>(
            qb, uT, a, nullptr,
            Tc, Dc, Kc,
            Kc, Kc, Dc, 1, 0,
            (long long)Hc * Tc * Kc, (long long)Tc * Kc,
            (long long)Hc * Kc * Dc, (long long)Kc * Dc,
            (long long)Hc * Tc * Dc, (long long)Tc * Dc, 0, 0, Hc);

        ln_rows_bf16<<<Bc * Hc * Tc, 256, 0, stream>>>(a, lab);

        // y = relu(ln(a) @ Dy[h]) * x -> bf16 written into [B,T,N] layout
        wmma_gemm<3><<<dim3(Kc / BNt, Tc / BMt, Bc * Hc), 256, 0, stream>>>(
            lab, DyT, yb, x,
            Tc, Kc, Dc,
            Dc, Dc, Nc, 1, Kc,
            (long long)Hc * Tc * Dc, (long long)Tc * Dc, 0, (long long)Dc * Kc,
            (long long)Tc * Nc, (long long)Kc,
            (long long)Hc * Tc * Kc, (long long)Tc * Kc, Hc);

        // w = y @ E : f32 out [M,D]   (B operand is ET[d][n])
        wmma_gemm<0><<<dim3(Dc / BNt, Mc / BMt, 1), 256, 0, stream>>>(
            yb, ET, w, nullptr,
            Mc, Dc, Nc,
            Nc, Nc, Dc, 1, 0,
            0, 0, 0, 0, 0, 0, 0, 0, 1);

        residual_ln<<<Mc, 256, 0, stream>>>(w, v);
    }

    // out = v @ readout   (B operand is roT[v][d])
    cvt_f32_bf16<<<(Mc * Dc) / 256, 256, 0, stream>>>(v, vp, Mc * Dc);
    wmma_gemm<0><<<dim3(Vc / BNt, Mc / BMt, 1), 256, 0, stream>>>(
        vp, roT, out, nullptr,
        Mc, Vc, Dc,
        Dc, Dc, Vc, 1, 0,
        0, 0, 0, 0, 0, 0, 0, 0, 1);
}
